// StockLSTM_21869973471394
// MI455X (gfx1250) — matrix operand order repaired and checked by reading, not compiled, not run on hardware
//
#include <hip/hip_runtime.h>
#include <hip/hip_bf16.h>

// ---------------------------------------------------------------------------
// 2-layer LSTM (B=256, T=512, I=64, H=512) + FC, for gfx1250 (MI455X).
// bf16 WMMA (v_wmma_f32_16x16x32_bf16) for every GEMM; weights pre-packed in
// a contiguous-k (k = 16h + j) B-fragment lane layout; gate nonlinearity fused via LDS.
// One kernel per (layer, timestep) acts as the grid-wide sync for the
// sequential recurrence. Weights (~6.3 MB bf16) are L2-resident (192 MB), so
// each step is latency-bound: wave tiles are 2Mx2N with explicit ping-pong
// register double-buffering so 4 independent WMMAs overlap fragment loads.
// ---------------------------------------------------------------------------

typedef __bf16 bf16_t;
typedef __attribute__((ext_vector_type(16))) __bf16 v16bf;
typedef __attribute__((ext_vector_type(8)))  float  v8f;

union AB32 {          // one 32-byte WMMA operand fragment per lane
    v16bf v;
    uint4 u[2];
};

#define HSZ   512
#define FOURH 2048
#define BATCH 256
#define SEQ   512
#define ISZ   64

// ---------------------------------------------------------------------------
// fp32 -> bf16 elementwise convert (for x)
// ---------------------------------------------------------------------------
__global__ __launch_bounds__(256) void cvt_f32_bf16_kernel(
    const float* __restrict__ src, bf16_t* __restrict__ dst, int n)
{
    int i = blockIdx.x * 256 + threadIdx.x;
    if (i < n) dst[i] = (bf16_t)src[i];
}

// ---------------------------------------------------------------------------
// Pack W [2048, K] (row-major fp32, gates i,f,g,o stacked) into WMMA
// B-fragments (bf16). Fragment (nt, kc) holds the 32x16 (KxN) tile:
//   lane l<16 : N=l,       halves j=0..15 -> K = kc*32 + j
//   lane l>=16: N=l-16,    halves j=0..15 -> K = kc*32 + 16 + j
// Flat out index = ((nt*nKc + kc)*32 + lane)*16 + j.
// ---------------------------------------------------------------------------
__global__ __launch_bounds__(256) void pack_weight_kernel(
    const float* __restrict__ W, bf16_t* __restrict__ out, int K)
{
    int idx = blockIdx.x * 256 + threadIdx.x;
    int total = FOURH * K;
    if (idx >= total) return;
    int nKc  = K >> 5;
    int j    = idx & 15;
    int lane = (idx >> 4) & 31;
    int rem  = idx >> 9;
    int kc   = rem % nKc;
    int nt   = rem / nKc;
    int n    = lane & 15;
    int k    = kc * 32 + ((lane >> 4) << 4) + j;
    int grow = nt * 16 + n;
    out[idx] = (bf16_t)W[(size_t)grow * K + k];
}

// ---------------------------------------------------------------------------
// One LSTM step for one layer, fused GEMM + cell update.
// Grid: (BATCH/32, HSZ/64) WGs of 256 threads (8 waves).
// Wave w: gate = w>>1, column-half nh = w&1 -> 32 cols (two 16-col N tiles),
// two 16-row M tiles => 4 accumulators, 4 WMMAs per 32-K chunk.
// ---------------------------------------------------------------------------
__global__ __launch_bounds__(256) void lstm_step_kernel(
    const bf16_t* __restrict__ x_in, int x_row_stride, int K_in,
    const bf16_t* __restrict__ h_prev,
    const bf16_t* __restrict__ Wih_pack,
    const bf16_t* __restrict__ Whh_pack,
    const float*  __restrict__ b_ih,
    const float*  __restrict__ b_hh,
    float*        __restrict__ c_state,
    bf16_t*       __restrict__ h_out)
{
    __shared__ float gbuf[4][64][33];   // [gate][n][m], padded vs bank conflicts

    const int wave    = threadIdx.x >> 5;
    const int lane    = threadIdx.x & 31;
    const int gate    = wave >> 1;
    const int nh      = wave & 1;
    const int nLocal  = lane & 15;
    const int hiHalf  = lane >> 4;
    const int rowBase = blockIdx.x * 32;        // batch rows
    const int colWG   = blockIdx.y * 64;        // hidden cols per WG

    const int ct0  = (colWG >> 4) + nh * 2;     // 16-col tile index in [0,32)
    const int nt0  = gate * 32 + ct0;           // B-fragment tile ids
    const int nt1  = nt0 + 1;
    const int col0 = ct0 * 16 + nLocal;         // hidden col of this lane's N
    const int col1 = col0 + 16;

    // Bias is per-column (per-N); each lane's C fragment has fixed N, so the
    // bias folds into the accumulator init for free.
    const float bias0 = b_ih[gate * HSZ + col0] + b_hh[gate * HSZ + col0];
    const float bias1 = b_ih[gate * HSZ + col1] + b_hh[gate * HSZ + col1];
    v8f acc00, acc01, acc10, acc11;             // [mtile][ntile]
    #pragma unroll
    for (int r = 0; r < 8; ++r) {
        acc00[r] = bias0; acc01[r] = bias1;
        acc10[r] = bias0; acc11[r] = bias1;
    }

    // ---- K-accumulation over one operand segment (nKc is even: 2 or 16) ----
    auto mma_seg = [&](const bf16_t* __restrict__ src, int rstride,
                       const bf16_t* __restrict__ Wp, int Kseg) {
        const int nKc = Kseg >> 5;
        const bf16_t* w0 = Wp + ((size_t)nt0 * nKc) * 512 + lane * 16;
        const bf16_t* w1 = Wp + ((size_t)nt1 * nKc) * 512 + lane * 16;
        const bf16_t* a0 = src + (size_t)(rowBase +  0 + nLocal) * rstride + hiHalf * 16;
        const bf16_t* a1 = src + (size_t)(rowBase + 16 + nLocal) * rstride + hiHalf * 16;

        AB32 A0p, A1p, B0p, B1p;    // ping buffers
        AB32 A0q, A1q, B0q, B1q;    // pong buffers

        auto loadA = [&](AB32& X0, AB32& X1, int kb) {
            // A layout used here (matches packed B): lo lanes K {0..15}, hi lanes {16..31}
            X0.u[0] = *(const uint4*)(a0 + kb);
            X0.u[1] = *(const uint4*)(a0 + kb + 8);
            X1.u[0] = *(const uint4*)(a1 + kb);
            X1.u[1] = *(const uint4*)(a1 + kb + 8);
        };
        auto loadB = [&](AB32& X0, AB32& X1, int kc) {
            const bf16_t* p0 = w0 + (size_t)kc * 512;
            const bf16_t* p1 = w1 + (size_t)kc * 512;
            X0.u[0] = *(const uint4*)(p0);
            X0.u[1] = *(const uint4*)(p0 + 8);
            X1.u[0] = *(const uint4*)(p1);
            X1.u[1] = *(const uint4*)(p1 + 8);
        };
        auto mma4 = [&](const AB32& Am0, const AB32& Am1,
                        const AB32& Bn0, const AB32& Bn1) {
            acc00 = __builtin_amdgcn_wmma_f32_16x16x32_bf16(
                        false, Am0.v, false, Bn0.v, (short)0, acc00, false, false);
            acc10 = __builtin_amdgcn_wmma_f32_16x16x32_bf16(
                        false, Am1.v, false, Bn0.v, (short)0, acc10, false, false);
            acc01 = __builtin_amdgcn_wmma_f32_16x16x32_bf16(
                        false, Am0.v, false, Bn1.v, (short)0, acc01, false, false);
            acc11 = __builtin_amdgcn_wmma_f32_16x16x32_bf16(
                        false, Am1.v, false, Bn1.v, (short)0, acc11, false, false);
        };

        // Explicit software pipeline: prefetch chunk k+1 before math on k.
        loadA(A0p, A1p, 0);
        loadB(B0p, B1p, 0);
        for (int kc = 0; kc < nKc; kc += 2) {
            loadA(A0q, A1q, (kc + 1) * 32);
            loadB(B0q, B1q, kc + 1);
            mma4(A0p, A1p, B0p, B1p);
            if (kc + 2 < nKc) {
                loadA(A0p, A1p, (kc + 2) * 32);
                loadB(B0p, B1p, kc + 2);
            }
            mma4(A0q, A1q, B0q, B1q);
        }
    };

    mma_seg(x_in,   x_row_stride, Wih_pack, K_in);   // input projection
    mma_seg(h_prev, HSZ,          Whh_pack, HSZ);    // recurrent projection

    // ---- spill C fragments to LDS: C layout VGPR r -> M=r (+8 hi lanes) ----
    {
        const int m0 = hiHalf * 8;
        const int nA = nh * 32 + nLocal;       // local col of N tile 0
        const int nB = nA + 16;                // local col of N tile 1
        #pragma unroll
        for (int r = 0; r < 8; ++r) {
            gbuf[gate][nA][m0 + r]      = acc00[r];
            gbuf[gate][nA][16 + m0 + r] = acc10[r];
            gbuf[gate][nB][m0 + r]      = acc01[r];
            gbuf[gate][nB][16 + m0 + r] = acc11[r];
        }
    }
    __syncthreads();

    // ---- fused LSTM cell update: 32 rows x 64 cols per WG ----
    for (int e = threadIdx.x; e < 32 * 64; e += 256) {
        const int m = e & 31;
        const int n = e >> 5;
        float iv = gbuf[0][n][m];
        float fv = gbuf[1][n][m];
        float gv = gbuf[2][n][m];
        float ov = gbuf[3][n][m];
        iv = 1.0f / (1.0f + __expf(-iv));
        fv = 1.0f / (1.0f + __expf(-fv));
        ov = 1.0f / (1.0f + __expf(-ov));
        gv = tanhf(gv);
        const size_t off = (size_t)(rowBase + m) * HSZ + (colWG + n);
        const float c = fv * c_state[off] + iv * gv;
        c_state[off] = c;
        h_out[off]   = (bf16_t)(ov * tanhf(c));
    }
}

// ---------------------------------------------------------------------------
// Final FC: out[b] = h_last[b,:] . fc_w + fc_b   (tiny; one WG per row)
// ---------------------------------------------------------------------------
__global__ __launch_bounds__(256) void fc_kernel(
    const bf16_t* __restrict__ h, const float* __restrict__ fc_w,
    const float* __restrict__ fc_b, float* __restrict__ out)
{
    const int b = blockIdx.x;
    float s = 0.0f;
    for (int i = threadIdx.x; i < HSZ; i += 256)
        s += (float)h[(size_t)b * HSZ + i] * fc_w[i];
    #pragma unroll
    for (int off = 16; off > 0; off >>= 1)
        s += __shfl_down(s, off, 32);
    __shared__ float red[8];
    if ((threadIdx.x & 31) == 0) red[threadIdx.x >> 5] = s;
    __syncthreads();
    if (threadIdx.x == 0) {
        float t = 0.0f;
        #pragma unroll
        for (int i = 0; i < 8; ++i) t += red[i];
        out[b] = t + fc_b[0];
    }
}

// ---------------------------------------------------------------------------
// Host orchestration
// ---------------------------------------------------------------------------
extern "C" void kernel_launch(void* const* d_in, const int* in_sizes, int n_in,
                              void* d_out, int out_size, void* d_ws, size_t ws_size,
                              hipStream_t stream)
{
    (void)in_sizes; (void)n_in; (void)out_size; (void)ws_size;

    const float* x     = (const float*)d_in[0];
    const float* W_ih0 = (const float*)d_in[1];
    const float* W_hh0 = (const float*)d_in[2];
    const float* b_ih0 = (const float*)d_in[3];
    const float* b_hh0 = (const float*)d_in[4];
    const float* W_ih1 = (const float*)d_in[5];
    const float* W_hh1 = (const float*)d_in[6];
    const float* b_ih1 = (const float*)d_in[7];
    const float* b_hh1 = (const float*)d_in[8];
    const float* fc_w  = (const float*)d_in[9];
    const float* fc_b  = (const float*)d_in[10];
    float* out = (float*)d_out;

    char* ws = (char*)d_ws;
    // ---- workspace layout (all offsets 256B aligned) ----
    const size_t off_xbf   = 0;                                     // 16 MB
    const size_t off_w0ih  = off_xbf  + (size_t)BATCH*SEQ*ISZ*2;    // 2048*64  bf16
    const size_t off_w0hh  = off_w0ih + (size_t)FOURH*ISZ*2;        // 2048*512 bf16
    const size_t off_w1ih  = off_w0hh + (size_t)FOURH*HSZ*2;
    const size_t off_w1hh  = off_w1ih + (size_t)FOURH*HSZ*2;
    const size_t off_state = off_w1hh + (size_t)FOURH*HSZ*2;
    // state region: c0, c1 (fp32), h0[2], h1[2] (bf16), each BATCH*HSZ elems
    const size_t n_st = (size_t)BATCH * HSZ;
    float*  c0  = (float*) (ws + off_state);
    float*  c1  = (float*) (ws + off_state + n_st*4);
    bf16_t* h0a = (bf16_t*)(ws + off_state + n_st*8);
    bf16_t* h0b = (bf16_t*)(ws + off_state + n_st*8 + n_st*2);
    bf16_t* h1a = (bf16_t*)(ws + off_state + n_st*8 + n_st*4);
    bf16_t* h1b = (bf16_t*)(ws + off_state + n_st*8 + n_st*6);
    const size_t state_bytes = n_st*16;          // 2 MB total

    bf16_t* x_bf = (bf16_t*)(ws + off_xbf);
    bf16_t* w0ih = (bf16_t*)(ws + off_w0ih);
    bf16_t* w0hh = (bf16_t*)(ws + off_w0hh);
    bf16_t* w1ih = (bf16_t*)(ws + off_w1ih);
    bf16_t* w1hh = (bf16_t*)(ws + off_w1hh);

    // ---- per-call prep (deterministic; no static state) ----
    {
        const int nx = BATCH * SEQ * ISZ;
        cvt_f32_bf16_kernel<<<(nx + 255) / 256, 256, 0, stream>>>(x, x_bf, nx);
        pack_weight_kernel<<<(FOURH * ISZ + 255) / 256, 256, 0, stream>>>(W_ih0, w0ih, ISZ);
        pack_weight_kernel<<<(FOURH * HSZ + 255) / 256, 256, 0, stream>>>(W_hh0, w0hh, HSZ);
        pack_weight_kernel<<<(FOURH * HSZ + 255) / 256, 256, 0, stream>>>(W_ih1, w1ih, HSZ);
        pack_weight_kernel<<<(FOURH * HSZ + 255) / 256, 256, 0, stream>>>(W_hh1, w1hh, HSZ);
        hipMemsetAsync(ws + off_state, 0, state_bytes, stream);  // h = c = 0
    }

    bf16_t* h0buf[2] = { h0a, h0b };
    bf16_t* h1buf[2] = { h1a, h1b };

    const dim3 grid(BATCH / 32, HSZ / 64);   // 8 x 8 WGs, 256 threads each
    for (int t = 0; t < SEQ; ++t) {
        const int rp = t & 1, wp = 1 - rp;
        // layer 0: input row b of x_t is x_bf + b*T*I + t*I
        lstm_step_kernel<<<grid, 256, 0, stream>>>(
            x_bf + (size_t)t * ISZ, SEQ * ISZ, ISZ,
            h0buf[rp], w0ih, w0hh, b_ih0, b_hh0, c0, h0buf[wp]);
        // layer 1: input = h0_t just produced
        lstm_step_kernel<<<grid, 256, 0, stream>>>(
            h0buf[wp], HSZ, HSZ,
            h1buf[rp], w1ih, w1hh, b_ih1, b_hh1, c1, h1buf[wp]);
    }
    // after t = SEQ-1 (odd), final h1 lives in parity 0
    fc_kernel<<<BATCH, 256, 0, stream>>>(h1buf[0], fc_w, fc_b, out);
}
